// FixedAttention_25520695673350
// MI455X (gfx1250) — matrix operand order repaired and checked by reading, not compiled, not run on hardware
//
#include <hip/hip_runtime.h>

#define TCTX  2048
#define DHEAD 64
#define NHEAD 16
#define EMB   1024
#define NEGF  (-1e30f)

typedef _Float16 v16h __attribute__((ext_vector_type(16)));
typedef float    v8f  __attribute__((ext_vector_type(8)));

__device__ __forceinline__ v8f wmma16(v16h a, v16h b, v8f c) {
  // D = A(16x32 f16) * B(32x16 f16) + C(16x16 f32)
  return __builtin_amdgcn_wmma_f32_16x16x32_f16(false, a, false, b, (short)0, c, false, false);
}

// B operand: 16 contiguous halfs (col n = lane&15, K = 16*kh + i)
__device__ __forceinline__ v16h load_b16c(const _Float16* p) {
  v16h b;
#pragma unroll
  for (int i = 0; i < 16; ++i) b[i] = p[i];
  return b;
}

// A operand: row m = lane&15; WMMA K(i) = 16*kh + i, plus chunk base
__device__ __forceinline__ v16h load_a16(const _Float16* rowp, int kbase, int kh) {
  v16h a;
#pragma unroll
  for (int i = 0; i < 16; ++i) a[i] = rowp[kbase + 16 * kh + i];
  return a;
}

__global__ __launch_bounds__(128)
void FixedAttention_kernel(const float* __restrict__ Qg, const float* __restrict__ Kg,
                           const float* __restrict__ Vg, float* __restrict__ Og) {
  const int qb   = blockIdx.x;        // 128-row query block, 0..15
  const int h    = blockIdx.y;
  const int bi   = blockIdx.z;
  const int tid  = threadIdx.x;
  const int wv   = tid >> 5;          // wave 0..3, owns query rows [wv*32, wv*32+32)
  const int lane = tid & 31;
  const int ln16 = lane & 15;
  const int kh   = lane >> 4;

  __shared__ _Float16 Ks [128 * 72];   // K block, row-major (key, dhead), pad 72
  __shared__ _Float16 Vt [64 * 136];   // V block transposed (dhead, key), pad 136
  __shared__ _Float16 Ks2[16 * 72];    // gathered summary keys
  __shared__ _Float16 Vt2[64 * 40];    // gathered summary V, transposed, zero-padded K
  __shared__ _Float16 Ps [4][16 * 136];// per-wave P (16 x 128) for D->A transpose

  const size_t base = (size_t)bi * TCTX * EMB + (size_t)h * DHEAD;

  // Zero keys 16..31 of this wave's P buffer (summary pass runs a K=32 WMMA step).
  for (int i = lane; i < 16 * 16; i += 32)
    Ps[wv][(i >> 4) * 136 + 16 + (i & 15)] = (_Float16)0;

  // ---- Q as WMMA A operands, f32 -> f16, pre-scaled by 1/sqrt(64) ----
  v16h qa[2][2];
#pragma unroll
  for (int s = 0; s < 2; ++s) {
    const int qrow = qb * 128 + wv * 32 + s * 16 + ln16;
    const float* qp = Qg + base + (size_t)qrow * EMB;
#pragma unroll
    for (int st = 0; st < 2; ++st)
#pragma unroll
      for (int i = 0; i < 16; ++i) {
        const int kd = st * 32 + 16 * kh + i;
        qa[s][st][i] = (_Float16)(0.125f * qp[kd]);
      }
  }

  // ---- flash state ----
  float mrow[2][8], lrow[2][8];
  v8f acc[2][4];
#pragma unroll
  for (int s = 0; s < 2; ++s) {
#pragma unroll
    for (int r = 0; r < 8; ++r) { mrow[s][r] = NEGF; lrow[s][r] = 0.f; }
#pragma unroll
    for (int o = 0; o < 4; ++o)
#pragma unroll
      for (int r = 0; r < 8; ++r) acc[s][o][r] = 0.f;
  }

  const int nsum = qb - 1;   // fully-attended summary cols: c in [0, nsum)

  // ---- stage gathered summary K/V (keys j = 128c + 127) ----
  if (qb >= 2) {
    for (int i = tid; i < 64 * 40; i += 128) Vt2[i] = (_Float16)0;
  }
  __syncthreads();
  if (qb >= 2) {
    for (int i = tid; i < 256; i += 128) {     // 16 rows x 16 float4
      const int r  = i >> 4;
      const int c4 = (i & 15) * 4;
      float4 k4, v4;
      if (r < nsum) {
        const size_t g = base + (size_t)(r * 128 + 127) * EMB + c4;
        k4 = *reinterpret_cast<const float4*>(Kg + g);
        v4 = *reinterpret_cast<const float4*>(Vg + g);
      } else {
        k4.x = k4.y = k4.z = k4.w = 0.f; v4 = k4;
      }
      Ks2[r * 72 + c4 + 0] = (_Float16)k4.x;
      Ks2[r * 72 + c4 + 1] = (_Float16)k4.y;
      Ks2[r * 72 + c4 + 2] = (_Float16)k4.z;
      Ks2[r * 72 + c4 + 3] = (_Float16)k4.w;
      Vt2[(c4 + 0) * 40 + r] = (_Float16)v4.x;
      Vt2[(c4 + 1) * 40 + r] = (_Float16)v4.y;
      Vt2[(c4 + 2) * 40 + r] = (_Float16)v4.z;
      Vt2[(c4 + 3) * 40 + r] = (_Float16)v4.w;
    }
  }
  __syncthreads();

  // ---- summary pass: one 16-wide key tile ----
  if (qb >= 2) {
#pragma unroll
    for (int s = 0; s < 2; ++s) {
      v8f S;
#pragma unroll
      for (int r = 0; r < 8; ++r) S[r] = 0.f;
      const _Float16* bp = &Ks2[ln16 * 72 + 16 * kh];
      S = wmma16(qa[s][0], load_b16c(bp), S);
      S = wmma16(qa[s][1], load_b16c(bp + 32), S);
      const bool keep = (ln16 < nsum);
#pragma unroll
      for (int r = 0; r < 8; ++r) if (!keep) S[r] = NEGF;

      float mnew[8], al[8], rs[8];
#pragma unroll
      for (int r = 0; r < 8; ++r) {
        float v = S[r];
#pragma unroll
        for (int mk = 1; mk <= 8; mk <<= 1) v = fmaxf(v, __shfl_xor(v, mk));
        mnew[r] = fmaxf(mrow[s][r], v);
        al[r]   = __expf(mrow[s][r] - mnew[r]);
        mrow[s][r] = mnew[r];
      }
#pragma unroll
      for (int r = 0; r < 8; ++r) {
        float p = __expf(S[r] - mnew[r]);
        rs[r] = p;
        Ps[wv][(r + 8 * kh) * 136 + ln16] = (_Float16)p;
      }
#pragma unroll
      for (int r = 0; r < 8; ++r) {
        float v = rs[r];
#pragma unroll
        for (int mk = 1; mk <= 8; mk <<= 1) v += __shfl_xor(v, mk);
        lrow[s][r] = lrow[s][r] * al[r] + v;
#pragma unroll
        for (int o = 0; o < 4; ++o) acc[s][o][r] *= al[r];
      }
      v16h pa = load_a16(&Ps[wv][ln16 * 136], 0, kh);   // keys 16..31 are zeros
#pragma unroll
      for (int o = 0; o < 4; ++o) {
        const _Float16* vp = &Vt2[(o * 16 + ln16) * 40 + 16 * kh];
        acc[s][o] = wmma16(pa, load_b16c(vp), acc[s][o]);
      }
    }
  }

  // ---- local blocks: prev (strict-upper + col 127) and diag (causal lower) ----
  const int kb0 = (qb > 0) ? (qb - 1) : 0;
  for (int kb = kb0; kb <= qb; ++kb) {
    __syncthreads();
    {
      const float* kp = Kg + base + (size_t)(kb * 128) * EMB;
      const float* vp = Vg + base + (size_t)(kb * 128) * EMB;
      for (int i = tid; i < 2048; i += 128) {  // 128 rows x 16 float4
        const int r  = i >> 4;
        const int c4 = (i & 15) * 4;
        const float4 k4 = *reinterpret_cast<const float4*>(kp + (size_t)r * EMB + c4);
        const float4 v4 = *reinterpret_cast<const float4*>(vp + (size_t)r * EMB + c4);
        Ks[r * 72 + c4 + 0] = (_Float16)k4.x;
        Ks[r * 72 + c4 + 1] = (_Float16)k4.y;
        Ks[r * 72 + c4 + 2] = (_Float16)k4.z;
        Ks[r * 72 + c4 + 3] = (_Float16)k4.w;
        Vt[(c4 + 0) * 136 + r] = (_Float16)v4.x;
        Vt[(c4 + 1) * 136 + r] = (_Float16)v4.y;
        Vt[(c4 + 2) * 136 + r] = (_Float16)v4.z;
        Vt[(c4 + 3) * 136 + r] = (_Float16)v4.w;
      }
    }
    __syncthreads();

    const bool diag = (kb == qb);
#pragma unroll
    for (int s = 0; s < 2; ++s) {
      v8f st8[8];
#pragma unroll
      for (int nt = 0; nt < 8; ++nt) {
        v8f S;
#pragma unroll
        for (int r = 0; r < 8; ++r) S[r] = 0.f;
        const _Float16* bp = &Ks[(nt * 16 + ln16) * 72 + 16 * kh];
        S = wmma16(qa[s][0], load_b16c(bp), S);
        S = wmma16(qa[s][1], load_b16c(bp + 32), S);
        const int col = nt * 16 + ln16;
#pragma unroll
        for (int r = 0; r < 8; ++r) {
          const int row = wv * 32 + s * 16 + r + 8 * kh;
          const bool keep = diag ? (col <= row) : ((col > row) || (col == 127));
          if (!keep) S[r] = NEGF;
        }
        st8[nt] = S;
      }
      float mnew[8], al[8], rs[8];
#pragma unroll
      for (int r = 0; r < 8; ++r) {
        float v = st8[0][r];
#pragma unroll
        for (int nt = 1; nt < 8; ++nt) v = fmaxf(v, st8[nt][r]);
#pragma unroll
        for (int mk = 1; mk <= 8; mk <<= 1) v = fmaxf(v, __shfl_xor(v, mk));
        mnew[r] = fmaxf(mrow[s][r], v);
        al[r]   = __expf(mrow[s][r] - mnew[r]);
        mrow[s][r] = mnew[r];
        rs[r] = 0.f;
      }
#pragma unroll
      for (int nt = 0; nt < 8; ++nt)
#pragma unroll
        for (int r = 0; r < 8; ++r) {
          float p = __expf(st8[nt][r] - mnew[r]);
          rs[r] += p;
          Ps[wv][(r + 8 * kh) * 136 + nt * 16 + ln16] = (_Float16)p;
        }
#pragma unroll
      for (int r = 0; r < 8; ++r) {
        float v = rs[r];
#pragma unroll
        for (int mk = 1; mk <= 8; mk <<= 1) v += __shfl_xor(v, mk);
        lrow[s][r] = lrow[s][r] * al[r] + v;
#pragma unroll
        for (int o = 0; o < 4; ++o) acc[s][o][r] *= al[r];
      }
#pragma unroll
      for (int ks = 0; ks < 4; ++ks) {
        v16h pa = load_a16(&Ps[wv][ln16 * 136], ks * 32, kh);
#pragma unroll
        for (int o = 0; o < 4; ++o) {
          const _Float16* vp = &Vt[(o * 16 + ln16) * 136 + ks * 32 + 16 * kh];
          acc[s][o] = wmma16(pa, load_b16c(vp), acc[s][o]);
        }
      }
    }
  }

  // ---- epilogue: normalize and store f32 ----
#pragma unroll
  for (int s = 0; s < 2; ++s)
#pragma unroll
    for (int o = 0; o < 4; ++o)
#pragma unroll
      for (int r = 0; r < 8; ++r) {
        const int row = qb * 128 + wv * 32 + s * 16 + r + 8 * kh;
        const int col = o * 16 + ln16;
        Og[base + (size_t)row * EMB + col] = acc[s][o][r] / lrow[s][r];
      }
}

extern "C" void kernel_launch(void* const* d_in, const int* in_sizes, int n_in,
                              void* d_out, int out_size, void* d_ws, size_t ws_size,
                              hipStream_t stream) {
  (void)n_in; (void)out_size; (void)d_ws; (void)ws_size;
  const float* q = (const float*)d_in[0];
  const float* k = (const float*)d_in[1];
  const float* v = (const float*)d_in[2];
  float* out = (float*)d_out;
  const int b = in_sizes[0] / (TCTX * EMB);
  dim3 grid(TCTX / 128, NHEAD, b);
  hipLaunchKernelGGL(FixedAttention_kernel, grid, dim3(128), 0, stream, q, k, v, out);
}
